// ProtoDINO_47691316854830
// MI455X (gfx1250) — compile-verified, matrix operand order checked by reading, not run on hardware
//
#include <hip/hip_runtime.h>
#include <cstdint>
#include <cstddef>

// ---------------- problem constants ----------------
constexpr int cB    = 16;
constexpr int cNP   = 1296;
constexpr int cDIM  = 768;
constexpr int cC    = 201;
constexpr int cK    = 5;
constexpr int cDA   = 64;
constexpr int cN    = cB * cNP;          // 20736 patches
constexpr int cNCOL = cC * cK;           // 1005 prototype columns
constexpr int cNTIL = 63;                // ceil(1005/16)
constexpr int cMTIL = cN / 16;           // 1296 row tiles
constexpr float cINV_EPS = 20.0f;        // 1/0.05
constexpr float cGAMMA   = 0.999f;

typedef __attribute__((ext_vector_type(16))) _Float16 v16h;
typedef __attribute__((ext_vector_type(8)))  float    v8f;

// output offsets (floats) in d_out, reference tuple order
constexpr size_t OFF_CLS  = 0;
constexpr size_t OFF_L    = OFF_CLS + (size_t)cB * (cC - 1);              // 3200
constexpr size_t OFF_IMG  = OFF_L   + (size_t)cN * cNCOL;                  // + 20,839,680
constexpr size_t OFF_PAM  = OFF_IMG + (size_t)cB * cC * cK;
constexpr size_t OFF_PPL  = OFF_PAM + (size_t)cN;
constexpr size_t OFF_PNEW = OFF_PPL + (size_t)cN;

// ---------------- weight packing: wf [DIM,DA] -> B-fragment layout ----------------
// layout: [4 N-tiles][24 K-chunks][32 lanes][16 halves]
__global__ void k_pack_wfB(const float* __restrict__ wf, _Float16* __restrict__ wfB) {
    int idx = blockIdx.x * 256 + threadIdx.x;
    if (idx >= 4 * 24 * 32 * 16) return;
    int i    = idx & 15;
    int ln   = (idx >> 4) & 31;
    int rest = idx >> 9;
    int ch   = rest % 24;
    int t    = rest / 24;
    int k = ch * 32 + ((ln >> 4) << 4) + i;   // 0..767
    int n = t * 16 + (ln & 15);               // 0..63
    wfB[idx] = (_Float16)wf[(size_t)k * cDA + n];
}

// ---------------- prototype adapter: sigmoid(proto@wp+bp) then L2-norm ----------------
__global__ void k_proto_adapter(const float* __restrict__ protos,
                                const float* __restrict__ wp,
                                const float* __restrict__ bp,
                                float* __restrict__ protoAd) {
    int r = blockIdx.x;          // padded rows [0,1008)
    int j = threadIdx.x;         // 64 threads = adapter dim
    if (r >= cC * cK) { protoAd[(size_t)r * cDA + j] = 0.f; return; }
    __shared__ float prow[cDIM];
    for (int d = j; d < cDIM; d += 64) prow[d] = protos[(size_t)r * cDIM + d];
    __syncthreads();
    float dot = bp[j];
    for (int d = 0; d < cDIM; ++d) dot += prow[d] * wp[(size_t)d * cDA + j];
    float s = 1.f / (1.f + __expf(-dot));
    __shared__ float s2[64];
    s2[j] = s * s; __syncthreads();
    for (int st = 32; st > 0; st >>= 1) { if (j < st) s2[j] += s2[j + st]; __syncthreads(); }
    float nrm = fmaxf(sqrtf(s2[0]), 1e-12f);
    protoAd[(size_t)r * cDA + j] = s / nrm;
}

// pack proto_ad [1008,64] -> B-fragment layout: [63 tiles][2 chunks][32 lanes][16 halves]
__global__ void k_pack_protoB(const float* __restrict__ protoAd, _Float16* __restrict__ protoB) {
    int idx = blockIdx.x * 256 + threadIdx.x;
    if (idx >= cNTIL * 2 * 32 * 16) return;
    int i    = idx & 15;
    int ln   = (idx >> 4) & 31;
    int rest = idx >> 9;
    int ch   = rest & 1;
    int t    = rest >> 1;
    int d = ch * 32 + ((ln >> 4) << 4) + i;   // adapter dim 0..63
    int n = t * 16 + (ln & 15);               // column 0..1007
    protoB[idx] = (_Float16)protoAd[(size_t)n * cDA + d];
}

// ---------------- adapter GEMM: feat16 = f16(sigmoid(ptok @ wf + bf)) ----------------
// block = 128 (4 waves). Each wave owns a 16-row M strip, all 4 N-tiles (DA=64).
__global__ void __launch_bounds__(128) k_adapter(const float* __restrict__ ptok,
                                                 const _Float16* __restrict__ wfB,
                                                 const float* __restrict__ bf,
                                                 _Float16* __restrict__ feat16) {
    __shared__ __align__(32) _Float16 As[64 * 32];   // 64 rows x 32 k halves
    int m64  = blockIdx.x * 64;
    int w    = threadIdx.x >> 5;
    int lane = threadIdx.x & 31;
    int srow = threadIdx.x >> 1;
    int scol = (threadIdx.x & 1) * 16;
    v8f acc0 = {}, acc1 = {}, acc2 = {}, acc3 = {};
    for (int ch = 0; ch < 24; ++ch) {
        __syncthreads();
        const float* src = ptok + (size_t)(m64 + srow) * cDIM + ch * 32 + scol;
        v16h tmp;
#pragma unroll
        for (int i = 0; i < 16; ++i) tmp[i] = (_Float16)src[i];
        *(v16h*)&As[srow * 32 + scol] = tmp;
        if (ch + 1 < 24) __builtin_prefetch(src + 32, 0, 1);   // global_prefetch_b8
        __syncthreads();
        v16h a = *(const v16h*)&As[(w * 16 + (lane & 15)) * 32 + ((lane >> 4) << 4)];
        const _Float16* bbase = wfB + (size_t)ch * 512 + (size_t)lane * 16;
        v16h b0 = *(const v16h*)(bbase + 0 * 24 * 512);
        v16h b1 = *(const v16h*)(bbase + 1 * 24 * 512);
        v16h b2 = *(const v16h*)(bbase + 2 * 24 * 512);
        v16h b3 = *(const v16h*)(bbase + 3 * 24 * 512);
        acc0 = __builtin_amdgcn_wmma_f32_16x16x32_f16(false, a, false, b0, (short)0, acc0, false, false);
        acc1 = __builtin_amdgcn_wmma_f32_16x16x32_f16(false, a, false, b1, (short)0, acc1, false, false);
        acc2 = __builtin_amdgcn_wmma_f32_16x16x32_f16(false, a, false, b2, (short)0, acc2, false, false);
        acc3 = __builtin_amdgcn_wmma_f32_16x16x32_f16(false, a, false, b3, (short)0, acc3, false, false);
    }
    int colIn = lane & 15;
    int mrow  = m64 + w * 16 + (lane >> 4) * 8;
    float bv0 = bf[0 * 16 + colIn], bv1 = bf[1 * 16 + colIn];
    float bv2 = bf[2 * 16 + colIn], bv3 = bf[3 * 16 + colIn];
#pragma unroll
    for (int r = 0; r < 8; ++r) {
        size_t row = (size_t)(mrow + r) * cDA;
        feat16[row + 0 * 16 + colIn] = (_Float16)(1.f / (1.f + __expf(-(acc0[r] + bv0))));
        feat16[row + 1 * 16 + colIn] = (_Float16)(1.f / (1.f + __expf(-(acc1[r] + bv1))));
        feat16[row + 2 * 16 + colIn] = (_Float16)(1.f / (1.f + __expf(-(acc2[r] + bv2))));
        feat16[row + 3 * 16 + colIn] = (_Float16)(1.f / (1.f + __expf(-(acc3[r] + bv3))));
    }
}

// ---------------- main GEMM: L[20736,1005] = feat16 @ protoB^T ----------------
// block = 256 (8 waves). The 16x64 feat strip (2048 contiguous bytes) is staged into
// LDS with CDNA5 async-to-LDS DMA (GLOBAL_LOAD_ASYNC_TO_LDS_B128, tracked by ASYNCcnt):
// 128 lanes x 16B. Issuing waves drain ASYNCcnt, then the barrier publishes LDS.
__global__ void __launch_bounds__(256) k_gemm_L(const _Float16* __restrict__ feat16,
                                                const _Float16* __restrict__ protoB,
                                                float* __restrict__ Lout) {
    __shared__ __align__(32) _Float16 Fs[16 * 64];
    int mt   = blockIdx.x;
    int tid  = threadIdx.x;
    int w    = tid >> 5;
    int lane = tid & 31;
    if (tid < 128) {
        const _Float16* g = feat16 + (size_t)mt * 1024 + (size_t)tid * 8;  // 8 halves = 16B
        uint32_t ldsOff = (uint32_t)(uintptr_t)(&Fs[tid * 8]);             // low 32b of generic = LDS offset
        asm volatile("global_load_async_to_lds_b128 %0, %1, off"
                     :: "v"(ldsOff), "v"(g) : "memory");
        asm volatile("s_wait_asynccnt 0x0" ::: "memory");
    }
    __syncthreads();
    int nt  = blockIdx.y * 8 + w;                  // 0..63
    int ntc = nt < cNTIL ? nt : cNTIL - 1;         // clamp loads; EXEC stays full for WMMA
    v8f acc = {};
#pragma unroll
    for (int ch = 0; ch < 2; ++ch) {
        v16h a = *(const v16h*)&Fs[(lane & 15) * 64 + ch * 32 + ((lane >> 4) << 4)];
        v16h b = *(const v16h*)&protoB[(size_t)((ntc * 2 + ch) * 32 + lane) * 16];
        acc = __builtin_amdgcn_wmma_f32_16x16x32_f16(false, a, false, b, (short)0, acc, false, false);
    }
    if (nt < cNTIL) {
        int col = nt * 16 + (lane & 15);
        if (col < cNCOL) {
            int mb = mt * 16 + (lane >> 4) * 8;
#pragma unroll
            for (int r = 0; r < 8; ++r) Lout[(size_t)(mb + r) * cNCOL + col] = acc[r];
        }
    }
}

// ---------------- image_logits = max over NP ----------------
__global__ void k_imglogits(const float* __restrict__ Lmat, float* __restrict__ out) {
    int b = blockIdx.x / cC, c = blockIdx.x % cC;
    float mk[cK];
#pragma unroll
    for (int k = 0; k < cK; ++k) mk[k] = -3.4e38f;
    for (int p = threadIdx.x; p < cNP; p += 128) {
        const float* lp = Lmat + (size_t)(b * cNP + p) * cNCOL + c * cK;
#pragma unroll
        for (int k = 0; k < cK; ++k) mk[k] = fmaxf(mk[k], lp[k]);
    }
    __shared__ float sred[128];
    for (int k = 0; k < cK; ++k) {
        sred[threadIdx.x] = mk[k]; __syncthreads();
        for (int s = 64; s > 0; s >>= 1) {
            if (threadIdx.x < s) sred[threadIdx.x] = fmaxf(sred[threadIdx.x], sred[threadIdx.x + s]);
            __syncthreads();
        }
        if (threadIdx.x == 0) out[((size_t)b * cC + c) * cK + k] = sred[0];
        __syncthreads();
    }
}

// ---------------- class_logits = sum_k image_logits * softmax(sa)*K ----------------
__global__ void k_classlogits(const float* __restrict__ img, const float* __restrict__ sa,
                              float* __restrict__ out) {
    int b = blockIdx.x, c = threadIdx.x;
    if (c >= cC - 1) return;
    float v[cK], m = -3.4e38f;
#pragma unroll
    for (int k = 0; k < cK; ++k) { v[k] = sa[c * cK + k]; m = fmaxf(m, v[k]); }
    float s = 0.f;
#pragma unroll
    for (int k = 0; k < cK; ++k) { v[k] = __expf(v[k] - m); s += v[k]; }
    float r = 0.f;
#pragma unroll
    for (int k = 0; k < cK; ++k)
        r += img[((size_t)b * cC + c) * cK + k] * (v[k] / s * (float)cK);
    out[(size_t)b * (cC - 1) + c] = r;
}

__global__ void k_zero(float* __restrict__ imgsum) {
    if (threadIdx.x < cB * 2) imgsum[threadIdx.x] = 0.f;
}

// ---------------- per-patch L2 norms ----------------
__global__ void k_norms(const float* __restrict__ ptok, float* __restrict__ norms) {
    int n = blockIdx.x;
    float ss = 0.f;
    for (int d = threadIdx.x; d < cDIM; d += 128) {
        float x = ptok[(size_t)n * cDIM + d];
        ss += x * x;
    }
    __shared__ float sred[128];
    sred[threadIdx.x] = ss; __syncthreads();
    for (int s = 64; s > 0; s >>= 1) {
        if (threadIdx.x < s) sred[threadIdx.x] += sred[threadIdx.x + s];
        __syncthreads();
    }
    if (threadIdx.x == 0) norms[n] = fmaxf(sqrtf(sred[0]), 1e-12f);
}

// ---------------- fg/bg similarity + per-image L1 sums ----------------
__global__ void k_fgbg(const float* __restrict__ Lmat, const int* __restrict__ labels,
                       float* __restrict__ fg, float* __restrict__ bg, float* __restrict__ imgsum) {
    int n = blockIdx.x * 256 + threadIdx.x;
    if (n >= cN) return;
    int b = n / cNP;
    int lab = labels[b];
    const float* lp = Lmat + (size_t)n * cNCOL;
    float f = 0.f, g = 0.f;
#pragma unroll
    for (int k = 0; k < cK; ++k) { f += lp[lab * cK + k]; g += lp[(cC - 1) * cK + k]; }
    fg[n] = f; bg[n] = g;
    atomicAdd(&imgsum[b * 2 + 0], fabsf(f));
    atomicAdd(&imgsum[b * 2 + 1], fabsf(g));
}

__global__ void k_plabel(const float* __restrict__ fg, const float* __restrict__ bg,
                         const float* __restrict__ imgsum, const int* __restrict__ labels,
                         int* __restrict__ plab, float* __restrict__ outPpl) {
    int n = blockIdx.x * 256 + threadIdx.x;
    if (n >= cN) return;
    int b = n / cNP;
    float fn = fg[n] / fmaxf(imgsum[b * 2 + 0], 1e-12f);
    float gn = bg[n] / fmaxf(imgsum[b * 2 + 1], 1e-12f);
    int pl = (fn > gn) ? labels[b] : (cC - 1);
    plab[n] = pl;
    outPpl[n] = (float)pl;
}

// ---------------- masked sinkhorn: one block per class ----------------
__global__ void __launch_bounds__(256) k_sinkhorn(const float* __restrict__ Lmat,
                                                  const int* __restrict__ plab,
                                                  float* __restrict__ qown,
                                                  float* __restrict__ outPam) {
    int c = blockIdx.x, tid = threadIdx.x;
    __shared__ float sred[256];
    __shared__ float cs[cK];

    // pass 1: member count + masked max
    float lmx = -3.4e38f, cnt = 0.f;
    for (int n = tid; n < cN; n += 256) {
        if (plab[n] == c) {
            cnt += 1.f;
            const float* lp = Lmat + (size_t)n * cNCOL + c * cK;
#pragma unroll
            for (int k = 0; k < cK; ++k) lmx = fmaxf(lmx, lp[k]);
        }
    }
    sred[tid] = cnt; __syncthreads();
    for (int s = 128; s > 0; s >>= 1) { if (tid < s) sred[tid] += sred[tid + s]; __syncthreads(); }
    float ncf = sred[0]; __syncthreads();
    sred[tid] = lmx; __syncthreads();
    for (int s = 128; s > 0; s >>= 1) { if (tid < s) sred[tid] = fmaxf(sred[tid], sred[tid + s]); __syncthreads(); }
    lmx = sred[0]; __syncthreads();
    if (ncf < 0.5f) lmx = 0.f;

    // pass 2: Q = exp((L - lmax)/EPS)  (global scale cancels in first col-norm)
    for (int n = tid; n < cN; n += 256) {
        if (plab[n] == c) {
            const float* lp = Lmat + (size_t)n * cNCOL + c * cK;
            float* q = qown + (size_t)n * cK;
#pragma unroll
            for (int k = 0; k < cK; ++k) q[k] = __expf((lp[k] - lmx) * cINV_EPS);
        }
    }
    __syncthreads();

    // 3 sinkhorn iterations: col-norm (/K) then row-norm (/n_c)
    for (int it = 0; it < 3; ++it) {
        float lc[cK] = {0.f, 0.f, 0.f, 0.f, 0.f};
        for (int n = tid; n < cN; n += 256) {
            if (plab[n] == c) {
                const float* q = qown + (size_t)n * cK;
#pragma unroll
                for (int k = 0; k < cK; ++k) lc[k] += q[k];
            }
        }
        for (int k = 0; k < cK; ++k) {
            sred[tid] = lc[k]; __syncthreads();
            for (int s = 128; s > 0; s >>= 1) { if (tid < s) sred[tid] += sred[tid + s]; __syncthreads(); }
            if (tid == 0) cs[k] = sred[0];
            __syncthreads();
        }
        float ncg = fmaxf(ncf, 1.f);
        for (int n = tid; n < cN; n += 256) {
            if (plab[n] == c) {
                float* q = qown + (size_t)n * cK;
                float t[cK], rs = 0.f;
#pragma unroll
                for (int k = 0; k < cK; ++k) { t[k] = q[k] / fmaxf(cs[k], 1e-12f) * 0.2f; rs += t[k]; }
                float inv = 1.f / (fmaxf(rs, 1e-12f) * ncg);
#pragma unroll
                for (int k = 0; k < cK; ++k) q[k] = t[k] * inv;
            }
        }
        __syncthreads();
    }

    // final: Q *= n_c ; part assignment = argmax_k + c*K
    for (int n = tid; n < cN; n += 256) {
        if (plab[n] == c) {
            float* q = qown + (size_t)n * cK;
            float best = -1.f; int bi = 0;
#pragma unroll
            for (int k = 0; k < cK; ++k) {
                float v = q[k] * ncf; q[k] = v;
                if (v > best) { best = v; bi = k; }
            }
            outPam[n] = (float)(bi + c * cK);
        }
    }
}

// ---------------- prototype EMA update: one block per (class, part) ----------------
__global__ void __launch_bounds__(256) k_pnew(const float* __restrict__ ptok,
                                              const float* __restrict__ protos,
                                              const int* __restrict__ plab,
                                              const float* __restrict__ qown,
                                              const float* __restrict__ norms,
                                              float* __restrict__ out) {
    int c = blockIdx.x, k = blockIdx.y;
    __shared__ float acc[cDIM];
    for (int d = threadIdx.x; d < cDIM; d += 256) acc[d] = 0.f;
    __syncthreads();
    int cnt = 0;
    for (int n = 0; n < cN; ++n) {
        if (plab[n] == c) {
            ++cnt;
            float q = qown[(size_t)n * cK + k] / norms[n];
            const float* pr = ptok + (size_t)n * cDIM;
            for (int d = threadIdx.x; d < cDIM; d += 256) acc[d] += q * pr[d];
        }
    }
    __syncthreads();
    size_t base = ((size_t)c * cK + k) * cDIM;
    for (int d = threadIdx.x; d < cDIM; d += 256) {
        float p = protos[base + d];
        out[base + d] = (cnt > 0) ? (cGAMMA * p + (1.f - cGAMMA) * acc[d]) : p;
    }
}

// ---------------- launcher ----------------
extern "C" void kernel_launch(void* const* d_in, const int* in_sizes, int n_in,
                              void* d_out, int out_size, void* d_ws, size_t ws_size,
                              hipStream_t stream) {
    (void)in_sizes; (void)n_in; (void)out_size; (void)ws_size;
    const float* ptok   = (const float*)d_in[0];
    const float* protos = (const float*)d_in[1];
    const float* sa     = (const float*)d_in[2];
    const float* wf     = (const float*)d_in[3];
    const float* bf     = (const float*)d_in[4];
    const float* wp     = (const float*)d_in[5];
    const float* bp     = (const float*)d_in[6];
    const int*   labels = (const int*)d_in[7];
    float* out = (float*)d_out;
    char*  ws  = (char*)d_ws;

    size_t o = 0;
    auto alloc = [&](size_t bytes) { size_t r = o; o += (bytes + 255) & ~(size_t)255; return r; };
    _Float16* wfB    = (_Float16*)(ws + alloc((size_t)4 * 24 * 32 * 16 * sizeof(_Float16)));
    float*    protoAd= (float*)   (ws + alloc((size_t)1008 * cDA * sizeof(float)));
    _Float16* protoB = (_Float16*)(ws + alloc((size_t)cNTIL * 2 * 32 * 16 * sizeof(_Float16)));
    _Float16* feat16 = (_Float16*)(ws + alloc((size_t)cN * cDA * sizeof(_Float16)));
    float*    fg     = (float*)   (ws + alloc((size_t)cN * sizeof(float)));
    float*    bg     = (float*)   (ws + alloc((size_t)cN * sizeof(float)));
    float*    imgsum = (float*)   (ws + alloc((size_t)cB * 2 * sizeof(float)));
    int*      plab   = (int*)     (ws + alloc((size_t)cN * sizeof(int)));
    float*    qown   = (float*)   (ws + alloc((size_t)cN * cK * sizeof(float)));
    float*    norms  = (float*)   (ws + alloc((size_t)cN * sizeof(float)));

    float* outCls  = out + OFF_CLS;
    float* outL    = out + OFF_L;
    float* outImg  = out + OFF_IMG;
    float* outPam  = out + OFF_PAM;
    float* outPpl  = out + OFF_PPL;
    float* outPnew = out + OFF_PNEW;

    k_pack_wfB     <<<192, 256, 0, stream>>>(wf, wfB);
    k_proto_adapter<<<1008, 64, 0, stream>>>(protos, wp, bp, protoAd);
    k_pack_protoB  <<<252, 256, 0, stream>>>(protoAd, protoB);
    k_adapter      <<<cMTIL / 4, 128, 0, stream>>>(ptok, wfB, bf, feat16);
    k_gemm_L       <<<dim3(cMTIL, 8), 256, 0, stream>>>(feat16, protoB, outL);
    k_imglogits    <<<cB * cC, 128, 0, stream>>>(outL, outImg);
    k_classlogits  <<<cB, 256, 0, stream>>>(outImg, sa, outCls);
    k_zero         <<<1, 64, 0, stream>>>(imgsum);
    k_norms        <<<cN, 128, 0, stream>>>(ptok, norms);
    k_fgbg         <<<cN / 256, 256, 0, stream>>>(outL, labels, fg, bg, imgsum);
    k_plabel       <<<cN / 256, 256, 0, stream>>>(fg, bg, imgsum, labels, plab, outPpl);
    k_sinkhorn     <<<cC, 256, 0, stream>>>(outL, plab, qown, outPam);
    k_pnew         <<<dim3(cC, cK), 256, 0, stream>>>(ptok, protos, plab, qown, norms, outPnew);
}
